// TextureNet_86354612453945
// MI455X (gfx1250) — matrix operand order repaired and checked by reading, not compiled, not run on hardware
//
#include <hip/hip_runtime.h>

// ---------------------------------------------------------------------------
// Fused global-correlation softmax flow + self-attention propagation (unimatch)
// B=8, C=128, H=48, W=64, N=3072.  bf16 hi/lo split WMMA (3-product) for
// near-fp32 accuracy on v_wmma_f32_16x16x32_bf16, flash-style online softmax.
// B tiles staged through LDS with global_load_async_to_lds_b128 (ASYNCcnt),
// double-buffered, shared by the 8 waves of each workgroup.
// ---------------------------------------------------------------------------

typedef __attribute__((ext_vector_type(16))) __bf16 v16bf;
typedef __attribute__((ext_vector_type(8)))  float  v8f;

#define BB   8
#define CC   128
#define HH   48
#define WW   64
#define NN   3072        // H*W
#define NT   192         // NN / 16 column tiles
#define FRAG 512         // 32 lanes * 16 bf16 per fragment
#define TILE_BF (4*FRAG) // bf16 elems per 16-col K=128 tile (2048)
#define BLK_PER_B 24     // 192 M-tiles / 8 waves per block

static __device__ __forceinline__ v8f wmma_bf16(v16bf a, v16bf b, v8f c) {
  return __builtin_amdgcn_wmma_f32_16x16x32_bf16(
      /*neg_a=*/false, a, /*neg_b=*/false, b,
      /*c_mod=*/(short)0, c, /*reuse_a=*/false, /*reuse_b=*/false);
}

// async copy of 16 bytes global -> LDS (GV addressing, VDST = LDS byte offset)
static __device__ __forceinline__ void async_copy16(unsigned int lds_off,
                                                    const void* gaddr) {
  asm volatile("global_load_async_to_lds_b128 %0, %1, off"
               :: "v"(lds_off), "v"(gaddr) : "memory");
}

template <int N>
static __device__ __forceinline__ void wait_async() {
#if __has_builtin(__builtin_amdgcn_s_wait_asynccnt)
  __builtin_amdgcn_s_wait_asynccnt(N);
#else
  asm volatile("s_wait_asynccnt %0" :: "n"(N) : "memory");
#endif
}

// Load A-matrix fragments (16 rows x K=128) for one wave, splitting each fp32
// into bf16 hi + bf16 residual.  Source layout: src[c*NN + row]  ([C,N]).
// A-frag element map (contiguous K order, same as pack_b): lane half h = lane>>4,
// M = lane&15, elem i -> K = i + 16h.
static __device__ __forceinline__ void load_afrags(const float* __restrict__ src,
                                                   int row, int h,
                                                   v16bf* ah, v16bf* al) {
#pragma unroll
  for (int f = 0; f < 4; ++f) {
    v16bf vh, vl;
#pragma unroll
    for (int i = 0; i < 16; ++i) {
      int c = f * 32 + i + 16 * h;
      float v = src[(size_t)c * NN + row];
      __bf16 hb = (__bf16)v;
      vh[i] = hb;
      vl[i] = (__bf16)(v - (float)hb);
    }
    ah[f] = vh;
    al[f] = vl;
  }
}

// ---------------------------------------------------------------------------
// Repack a [C,N]-layout fp32 matrix into per-lane-contiguous bf16 (hi/lo)
// B-fragments: elem i of lane l (half h) holds K = f*32 + i + 16h for column
// ncol = t*16 + (l&15).  32-byte vector store per thread.
// ---------------------------------------------------------------------------
__global__ void pack_b(const float* __restrict__ src,
                       __bf16* __restrict__ dhi, __bf16* __restrict__ dlo) {
  int tid = blockIdx.x * blockDim.x + threadIdx.x;  // BB*NT*4*32 = 196608
  int l = tid & 31;
  int f = (tid >> 5) & 3;
  int t = (tid >> 7) % NT;
  int b = tid / (NT * 4 * 32);
  int h = l >> 4;
  int col = t * 16 + (l & 15);
  const float* sb = src + (size_t)b * CC * NN;
  v16bf vh, vl;
#pragma unroll
  for (int i = 0; i < 16; ++i) {
    int c = f * 32 + i + 16 * h;
    float v = sb[(size_t)c * NN + col];
    __bf16 hb = (__bf16)v;
    vh[i] = hb;
    vl[i] = (__bf16)(v - (float)hb);
  }
  size_t off = ((size_t)((b * NT + t) * 4 + f)) * FRAG + (size_t)l * 16;
  *(v16bf*)(dhi + off) = vh;
  *(v16bf*)(dlo + off) = vl;
}

// ---------------------------------------------------------------------------
// proj: out[co][n] = sum_c Asrc[c][n] * Wm[co][c] + bias[co]   (per batch)
// Asrc/out layout [B][C][N].  One wave per 16-row n-tile, 8 co-tiles, K=128.
// ---------------------------------------------------------------------------
__global__ void proj_gemm(const float* __restrict__ Asrc,
                          const float* __restrict__ Wm,
                          const float* __restrict__ bias,
                          float* __restrict__ outp) {
  int wave = (blockIdx.x * blockDim.x + threadIdx.x) >> 5;
  int lane = threadIdx.x & 31;
  int b  = wave / NT;
  int m0 = (wave % NT) * 16;
  int h   = lane >> 4;
  int l15 = lane & 15;

  const float* ab = Asrc + (size_t)b * CC * NN;
  v16bf ah[4], al[4];
  load_afrags(ab, m0 + l15, h, ah, al);

  float* ob = outp + (size_t)b * CC * NN;
#pragma unroll 1
  for (int cot = 0; cot < 8; ++cot) {
    int co = cot * 16 + l15;
    v8f acc = {};
#pragma unroll
    for (int f = 0; f < 4; ++f) {
      v16bf bh, bl;
#pragma unroll
      for (int i = 0; i < 16; ++i) {
        int c = f * 32 + i + 16 * h;
        float v = Wm[(size_t)co * CC + c];
        __bf16 hb = (__bf16)v;
        bh[i] = hb;
        bl[i] = (__bf16)(v - (float)hb);
      }
      acc = wmma_bf16(ah[f], bh, acc);
      acc = wmma_bf16(ah[f], bl, acc);
      acc = wmma_bf16(al[f], bh, acc);
    }
    float bv = bias[co];
#pragma unroll
    for (int r = 0; r < 8; ++r) {
      int n = m0 + r + 8 * h;
      ob[(size_t)co * NN + n] = acc[r] + bv;
    }
  }
}

// ---------------------------------------------------------------------------
// Flash kernel: rows of Asrc ([C,N] layout) against pre-packed B tiles staged
// in LDS via async copies.  Online softmax with value = grid coords (use_grid)
// or V[b][2][N].  Output out[b][2][N]; optionally subtract grid.
// 8 waves/block = 8 consecutive M-tiles of one batch; B tiles shared via LDS.
// ---------------------------------------------------------------------------
__global__ void flash_attn(const float* __restrict__ Asrc,
                           const __bf16* __restrict__ Bhi,
                           const __bf16* __restrict__ Blo,
                           const float* __restrict__ Vsrc,
                           float* __restrict__ outp,
                           int use_grid, int sub_grid, float scale) {
  __shared__ __bf16 smem[2][2][TILE_BF];   // [buffer][hi/lo][2048] = 16 KB

  int tid  = threadIdx.x;
  int lane = tid & 31;
  int b     = blockIdx.x / BLK_PER_B;
  int mtile = (blockIdx.x % BLK_PER_B) * 8 + (tid >> 5);
  int m0    = mtile * 16;
  int h   = lane >> 4;
  int l15 = lane & 15;

  v16bf ah[4], al[4];
  load_afrags(Asrc + (size_t)b * CC * NN, m0 + l15, h, ah, al);

  float mrun[8], srun[8], sx[8], sy[8];
#pragma unroll
  for (int r = 0; r < 8; ++r) { mrun[r] = -1e30f; srun[r] = 0.f; sx[r] = 0.f; sy[r] = 0.f; }

  const __bf16* bhi_b = Bhi + (size_t)b * NT * TILE_BF;
  const __bf16* blo_b = Blo + (size_t)b * NT * TILE_BF;
  const float*  vb    = Vsrc ? (Vsrc + (size_t)b * 2 * NN) : (const float*)0;

  // issue async copy of a whole packed tile (hi+lo, 8KB) into buffer `bufi`:
  // 256 threads x 4 x 16B  ->  4 async-load instructions per wave (ASYNCcnt +4)
  auto copy_tile = [&](int bufi, int t) {
    const __bf16* ghi = bhi_b + (size_t)t * TILE_BF;
    const __bf16* glo = blo_b + (size_t)t * TILE_BF;
#pragma unroll
    for (int j = 0; j < 2; ++j) {
      unsigned int off_e = ((unsigned)tid + 256u * j) * 8u;  // 8 bf16 = 16B
      async_copy16((unsigned int)(uintptr_t)&smem[bufi][0][off_e], ghi + off_e);
      async_copy16((unsigned int)(uintptr_t)&smem[bufi][1][off_e], glo + off_e);
    }
  };

  copy_tile(0, 0);

#pragma unroll 1
  for (int t = 0; t < NT; ++t) {
    int cur = t & 1;
    if (t + 1 < NT) {
      copy_tile(cur ^ 1, t + 1);
      wait_async<4>();           // tile t landed; tile t+1 still in flight
    } else {
      wait_async<0>();
    }
    __syncthreads();             // all waves' portions visible

    v8f acc = {};
    const __bf16* sh = &smem[cur][0][(size_t)lane * 16];
    const __bf16* sl = &smem[cur][1][(size_t)lane * 16];
#pragma unroll
    for (int f = 0; f < 4; ++f) {
      v16bf bh = *(const v16bf*)(sh + f * FRAG);
      v16bf bl = *(const v16bf*)(sl + f * FRAG);
      acc = wmma_bf16(ah[f], bh, acc);
      acc = wmma_bf16(ah[f], bl, acc);
      acc = wmma_bf16(al[f], bh, acc);
    }
    __syncthreads();             // done reading before buffer is re-filled

    int ncol = t * 16 + l15;
    float vx, vy;
    if (use_grid) { vx = (float)(ncol & (WW - 1)); vy = (float)(ncol >> 6); }
    else          { vx = vb[ncol];                 vy = vb[NN + ncol]; }
#pragma unroll
    for (int r = 0; r < 8; ++r) {
      float s  = acc[r] * scale;
      float nm = fmaxf(mrun[r], s);
      float cf = __expf(mrun[r] - nm);
      float p  = __expf(s - nm);
      srun[r] = srun[r] * cf + p;
      sx[r]   = sx[r]   * cf + p * vx;
      sy[r]   = sy[r]   * cf + p * vy;
      mrun[r] = nm;
    }
  }

  // merge partial softmax state across the 16 lanes of each half (wave32)
#pragma unroll
  for (int mask = 1; mask < 16; mask <<= 1) {
#pragma unroll
    for (int r = 0; r < 8; ++r) {
      float om = __shfl_xor(mrun[r], mask, 32);
      float os = __shfl_xor(srun[r], mask, 32);
      float ox = __shfl_xor(sx[r],   mask, 32);
      float oy = __shfl_xor(sy[r],   mask, 32);
      float nm = fmaxf(mrun[r], om);
      float a  = __expf(mrun[r] - nm);
      float c2 = __expf(om - nm);
      srun[r] = srun[r] * a + os * c2;
      sx[r]   = sx[r]   * a + ox * c2;
      sy[r]   = sy[r]   * a + oy * c2;
      mrun[r] = nm;
    }
  }

  if (l15 == 0) {  // lanes 0 and 16 write rows m0..m0+7 / m0+8..m0+15
    float* ob = outp + (size_t)b * 2 * NN;
#pragma unroll
    for (int r = 0; r < 8; ++r) {
      int n = m0 + r + 8 * h;
      float inv = 1.0f / srun[r];
      float fx = sx[r] * inv;
      float fy = sy[r] * inv;
      if (sub_grid) { fx -= (float)(n & (WW - 1)); fy -= (float)(n >> 6); }
      ob[n]      = fx;
      ob[NN + n] = fy;
    }
  }
}

// ---------------------------------------------------------------------------
extern "C" void kernel_launch(void* const* d_in, const int* in_sizes, int n_in,
                              void* d_out, int out_size, void* d_ws, size_t ws_size,
                              hipStream_t stream) {
  (void)in_sizes; (void)n_in; (void)out_size; (void)ws_size;
  const float* f0 = (const float*)d_in[0];
  const float* f1 = (const float*)d_in[1];
  const float* qw = (const float*)d_in[2];
  const float* qb = (const float*)d_in[3];
  const float* kw = (const float*)d_in[4];
  const float* kb = (const float*)d_in[5];

  float* outp      = (float*)d_out;
  float* flow      = outp;                 // [B,2,H,W]
  float* flow_pred = outp + BB * 2 * NN;   // [B,2,H,W]

  char* ws = (char*)d_ws;
  const size_t qt_bytes = (size_t)BB * CC * NN * sizeof(float);      // 12.6 MB
  float* q_t = (float*)ws;
  float* k_t = (float*)(ws + qt_bytes);
  __bf16* phi = (__bf16*)(ws + 2 * qt_bytes);
  __bf16* plo = phi + (size_t)BB * NT * TILE_BF;                     // 6.3 MB each

  const float scale = 0.0883883476483184f;  // 1/sqrt(128)

  const int packBlocks  = (BB * NT * 4 * 32) / 256;  // 768
  const int waveBlocks  = (BB * NT * 32) / 256;      // 192 (8 waves/block)

  // 1) global correlation softmax -> flow_pred
  pack_b<<<packBlocks, 256, 0, stream>>>(f1, phi, plo);
  flash_attn<<<waveBlocks, 256, 0, stream>>>(f0, phi, plo, (const float*)0,
                                             flow_pred, /*use_grid=*/1,
                                             /*sub_grid=*/1, scale);
  // 2) q = f0 @ qw^T + qb ; k = q @ kw^T + kb   (stored [C,N] per batch)
  proj_gemm<<<waveBlocks, 256, 0, stream>>>(f0,  qw, qb, q_t);
  proj_gemm<<<waveBlocks, 256, 0, stream>>>(q_t, kw, kb, k_t);
  // 3) self-attention over flow_pred -> flow
  pack_b<<<packBlocks, 256, 0, stream>>>(k_t, phi, plo);
  flash_attn<<<waveBlocks, 256, 0, stream>>>(q_t, phi, plo, flow_pred,
                                             flow, /*use_grid=*/0,
                                             /*sub_grid=*/0, scale);
}
